// AttentionHead_64802466562275
// MI455X (gfx1250) — matrix operand order repaired and checked by reading, not compiled, not run on hardware
//
#include <hip/hip_runtime.h>

typedef __attribute__((ext_vector_type(16))) _Float16 v16h;
typedef __attribute__((ext_vector_type(8)))  _Float16 v8h;
typedef __attribute__((ext_vector_type(8)))  float    v8f;

#define B_   4
#define T_   4096
#define E_   1024
#define DH_  128
#define NROW (B_ * T_)   // 16384

// ---------------------------------------------------------------------------
// Kernel 1: QKV projection.  grid = (NROW/16, 3), block = 256 (8 waves).
// Each block computes a 16-row x 128-col tile of Q, K, or V; wave w owns the
// 16-wide column slice [w*16, w*16+16).  K-dim (E=1024) streamed in chunks of
// 32 through LDS (W chunk stored transposed as f16 so B-fragments are
// contiguous ds_load_b128).  Q is pre-scaled by 1/sqrt(DH); V is stored
// TRANSPOSED ([b][dh][t]) so the attention kernel gets contiguous V loads.
// ---------------------------------------------------------------------------
__global__ __launch_bounds__(256)
void qkv_project_kernel(const float* __restrict__ X,
                        const float* __restrict__ Wq,
                        const float* __restrict__ Wk,
                        const float* __restrict__ Wv,
                        _Float16* __restrict__ Qh,
                        _Float16* __restrict__ Kh,
                        _Float16* __restrict__ Vt)
{
    __shared__ __attribute__((aligned(32))) _Float16 wsT[DH_ * 32]; // [n][k] 8KB
    __shared__ __attribute__((aligned(32))) _Float16 xs[16 * 32];   // [m][k] 1KB

    const int tid  = threadIdx.x;
    const int lane = tid & 31;
    const int w    = tid >> 5;          // wave id 0..7 -> N slice
    const int hi   = (lane >> 4) & 1;   // lane >= 16
    const int l15  = lane & 15;

    const int row0  = blockIdx.x * 16;  // flat row in [0, NROW)
    const int which = blockIdx.y;       // 0=Q 1=K 2=V
    const float* Wm = (which == 0) ? Wq : (which == 1) ? Wk : Wv;

    v8f acc = {};

    for (int k0 = 0; k0 < E_; k0 += 32) {
        __syncthreads();
        // Stage W[k0:k0+32, 0:128] transposed -> wsT[n*32+k] (f16).
        // 4096 elements / 256 threads = 16 each, coalesced along n.
        #pragma unroll
        for (int s = 0; s < 16; ++s) {
            int i  = tid + 256 * s;
            int kk = i >> 7;            // 0..31
            int nn = i & 127;           // 0..127
            wsT[nn * 32 + kk] = (_Float16)Wm[(size_t)(k0 + kk) * DH_ + nn];
        }
        // Stage X[row0:row0+16, k0:k0+32] -> xs[m*32+k] (2 elems/thread).
        {
            int mm  = tid >> 4;
            int kk0 = (tid & 15) * 2;
            const float* xr = X + (size_t)(row0 + mm) * E_ + k0 + kk0;
            xs[mm * 32 + kk0 + 0] = (_Float16)xr[0];
            xs[mm * 32 + kk0 + 1] = (_Float16)xr[1];
        }
        __syncthreads();

        // A fragment (16x32 f16): row m = l15; K = hi*16 + e (contiguous).
        v8h alo = *(const v8h*)&xs[l15 * 32 + hi * 16];
        v8h ahh = *(const v8h*)&xs[l15 * 32 + hi * 16 + 8];
        v16h a;
        #pragma unroll
        for (int i = 0; i < 8; ++i) { a[i] = alo[i]; a[i + 8] = ahh[i]; }

        // B fragment (32x16 f16): col n = w*16+l15; K = hi*16 + e (contiguous).
        v16h b = *(const v16h*)&wsT[(w * 16 + l15) * 32 + hi * 16];

        acc = __builtin_amdgcn_wmma_f32_16x16x32_f16(false, a, false, b,
                                                     (short)0, acc, false, false);
    }

    // C layout: VGPR r -> row m = r + hi*8, col n = w*16 + l15.
    // Uniform `which` hoisted out of the store loop -> three straight paths.
    const int n = w * 16 + l15;
    if (which == 0) {
        #pragma unroll
        for (int r = 0; r < 8; ++r) {
            int grow = row0 + r + hi * 8;
            Qh[(size_t)grow * DH_ + n] =
                (_Float16)(acc[r] * 0.08838834764831845f); // 1/sqrt(128)
        }
    } else if (which == 1) {
        #pragma unroll
        for (int r = 0; r < 8; ++r) {
            int grow = row0 + r + hi * 8;
            Kh[(size_t)grow * DH_ + n] = (_Float16)acc[r];
        }
    } else {
        #pragma unroll
        for (int r = 0; r < 8; ++r) {
            int grow = row0 + r + hi * 8;
            int bb = grow >> 12;               // / T_
            int tt = grow & (T_ - 1);          // % T_
            Vt[((size_t)bb * DH_ + n) * T_ + tt] = (_Float16)acc[r];
        }
    }
}

// ---------------------------------------------------------------------------
// Kernel 2: causal flash attention.  grid = NROW/64, block = 128 (4 waves).
// Each wave owns one 16-query tile and streams keys in chunks of 64 (four
// 16-key tiles) to amortize the softmax reductions and LDS transpose:
//   S = Q K^T        (16 wmma: 4 key tiles x 4 dh chunks)
//   causal mask + online softmax (16-lane shfl row reductions)
//   P -> LDS         (C->A layout transpose, wave-local, s_wait_dscnt ordered)
//   O = diag(a)O+PV  (16 wmma: 8 dh tiles x 2 K-halves, contiguous V^T loads)
// No block barriers: waves have different causal trip counts.
// ---------------------------------------------------------------------------
__global__ __launch_bounds__(128)
void flash_attn_kernel(const _Float16* __restrict__ Qh,
                       const _Float16* __restrict__ Kh,
                       const _Float16* __restrict__ Vt,
                       float* __restrict__ Out)
{
    // Per-wave P scratch: 16 rows x 64 keys f16 = 2KB; 4 waves = 8KB.
    __shared__ __attribute__((aligned(32))) _Float16 pbuf[4][16 * 64];

    const int tid  = threadIdx.x;
    const int lane = tid & 31;
    const int w    = tid >> 5;
    const int hi   = (lane >> 4) & 1;
    const int l15  = lane & 15;

    const int qtile = blockIdx.x * 4 + w;
    const int qrow0 = qtile * 16;          // flat row
    const int b     = qrow0 >> 12;         // / T_
    const int tq0   = qrow0 & (T_ - 1);    // within-batch query base

    // Q fragments (A layout), 4 dh-chunks of 32, kept in registers.
    v16h qf[4];
    {
        const _Float16* qr = Qh + (size_t)(qrow0 + l15) * DH_;
        #pragma unroll
        for (int c = 0; c < 4; ++c) {
            v8h lo = *(const v8h*)(qr + c * 32 + hi * 16);
            v8h hh = *(const v8h*)(qr + c * 32 + hi * 16 + 8);
            #pragma unroll
            for (int i = 0; i < 8; ++i) { qf[c][i] = lo[i]; qf[c][i + 8] = hh[i]; }
        }
    }

    v8f o[8] = {};
    float mrow[8], lrow[8];
    #pragma unroll
    for (int r = 0; r < 8; ++r) { mrow[r] = -1e30f; lrow[r] = 0.0f; }

    const _Float16* kb = Kh + (size_t)b * T_ * DH_;
    const _Float16* vb = Vt + (size_t)b * DH_ * T_;

    const int kend = tq0 + 16;             // last visible key (exclusive)
    for (int j = 0; j < kend; j += 64) {
        // Prefetch next key chunk (speculative; OOB silently dropped).
        __builtin_prefetch(kb + (size_t)(j + 64 + l15) * DH_, 0, 3);

        // ---- S = Q K^T for four 16-key tiles ----
        v8f s[4];
        #pragma unroll
        for (int t2 = 0; t2 < 4; ++t2) {
            v8f sacc = {};
            const _Float16* kr = kb + (size_t)(j + t2 * 16 + l15) * DH_;
            #pragma unroll
            for (int c = 0; c < 4; ++c) {
                v16h kf = *(const v16h*)(kr + c * 32 + hi * 16);
                sacc = __builtin_amdgcn_wmma_f32_16x16x32_f16(false, qf[c], false, kf,
                                                              (short)0, sacc, false, false);
            }
            s[t2] = sacc;
        }

        // ---- causal mask + online softmax (row stats via 16-lane shuffles) ----
        float alpha[8];
        #pragma unroll
        for (int r = 0; r < 8; ++r) {
            int tq  = tq0 + r + hi * 8;     // this element's query index
            int k0i = j + l15;              // this lane's key (tile 0)
            float sv[4];
            #pragma unroll
            for (int t2 = 0; t2 < 4; ++t2)
                sv[t2] = (k0i + 16 * t2 <= tq) ? s[t2][r] : -1e30f;
            float cm = fmaxf(fmaxf(sv[0], sv[1]), fmaxf(sv[2], sv[3]));
            cm = fmaxf(cm, __shfl_xor(cm, 1));
            cm = fmaxf(cm, __shfl_xor(cm, 2));
            cm = fmaxf(cm, __shfl_xor(cm, 4));
            cm = fmaxf(cm, __shfl_xor(cm, 8));
            float mnew = fmaxf(mrow[r], cm);
            float p[4], rs = 0.0f;
            #pragma unroll
            for (int t2 = 0; t2 < 4; ++t2) { p[t2] = __expf(sv[t2] - mnew); rs += p[t2]; }
            rs += __shfl_xor(rs, 1);
            rs += __shfl_xor(rs, 2);
            rs += __shfl_xor(rs, 4);
            rs += __shfl_xor(rs, 8);
            alpha[r] = __expf(mrow[r] - mnew);
            lrow[r]  = lrow[r] * alpha[r] + rs;
            mrow[r]  = mnew;
            // Spill P in C layout: row m = r + hi*8, cols l15 + 16*t2.
            int m = r + hi * 8;
            #pragma unroll
            for (int t2 = 0; t2 < 4; ++t2)
                pbuf[w][m * 64 + t2 * 16 + l15] = (_Float16)p[t2];
        }

        // Wave-local LDS ordering: DS ops from one wave complete in order.
        asm volatile("s_wait_dscnt 0" ::: "memory");

        // ---- reload P as two A fragments (16x32 each, key halves u) ----
        v16h pf[2];
        #pragma unroll
        for (int u = 0; u < 2; ++u) {
            const _Float16* pr = &pbuf[w][l15 * 64 + u * 32];
            v8h lo = *(const v8h*)(pr + hi * 16);
            v8h hh = *(const v8h*)(pr + hi * 16 + 8);
            #pragma unroll
            for (int i = 0; i < 8; ++i) { pf[u][i] = lo[i]; pf[u][i + 8] = hh[i]; }
        }

        // ---- O = diag(alpha) O + P V ----
        #pragma unroll
        for (int d = 0; d < 8; ++d) {
            #pragma unroll
            for (int r = 0; r < 8; ++r) o[d][r] *= alpha[r];
            const _Float16* vr = vb + (size_t)(d * 16 + l15) * T_ + j;
            #pragma unroll
            for (int u = 0; u < 2; ++u) {
                // B fragment: col n = d*16+l15 (dh), K = key j+32u+hi*16+e,
                // contiguous in transposed V.
                v16h vf = *(const v16h*)(vr + u * 32 + hi * 16);
                o[d] = __builtin_amdgcn_wmma_f32_16x16x32_f16(false, pf[u], false, vf,
                                                              (short)0, o[d], false, false);
            }
        }
    }

    // ---- normalize by row sums and store f32 output ----
    #pragma unroll
    for (int r = 0; r < 8; ++r) lrow[r] = 1.0f / lrow[r];
    #pragma unroll
    for (int d = 0; d < 8; ++d) {
        #pragma unroll
        for (int r = 0; r < 8; ++r) {
            int grow = qrow0 + r + hi * 8;
            Out[(size_t)grow * DH_ + d * 16 + l15] = o[d][r] * lrow[r];
        }
    }
}

// ---------------------------------------------------------------------------
extern "C" void kernel_launch(void* const* d_in, const int* in_sizes, int n_in,
                              void* d_out, int out_size, void* d_ws, size_t ws_size,
                              hipStream_t stream)
{
    (void)in_sizes; (void)n_in; (void)out_size; (void)ws_size;

    const float* X  = (const float*)d_in[0];
    const float* Wq = (const float*)d_in[1];
    const float* Wk = (const float*)d_in[2];
    const float* Wv = (const float*)d_in[3];

    // Workspace: Q/K/V in f16 (4 MB each) + 64 KB pad per region so masked
    // tail-chunk loads never run off the end.
    const size_t region = (size_t)NROW * DH_ * sizeof(_Float16) + 65536;
    char* ws = (char*)d_ws;
    _Float16* Qh = (_Float16*)(ws);
    _Float16* Kh = (_Float16*)(ws + region);
    _Float16* Vt = (_Float16*)(ws + 2 * region);

    dim3 g1(NROW / 16, 3, 1), b1(256, 1, 1);
    qkv_project_kernel<<<g1, b1, 0, stream>>>(X, Wq, Wk, Wv, Qh, Kh, Vt);

    dim3 g2(NROW / 64, 1, 1), b2(128, 1, 1);
    flash_attn_kernel<<<g2, b2, 0, stream>>>(Qh, Kh, Vt, (float*)d_out);
}
